// ConvMultiStepAttention_15410342658567
// MI455X (gfx1250) — matrix operand order repaired and checked by reading, not compiled, not run on hardware
//
#include <hip/hip_runtime.h>

// ---------------------------------------------------------------------------
// MI455X (gfx1250) fused conv-multistep-attention.
// bf16 WMMA (V_WMMA_F32_16X16X32_BF16) for all three GEMMs, f32 accumulate.
// All GEMM operands pre-transposed to K-contiguous layout so the hot loops use
// GLOBAL_LOAD_ASYNC_TO_LDS_B128 (ASYNCcnt) with double-buffered LDS staging.
// ---------------------------------------------------------------------------

typedef __bf16 bf16;
typedef bf16  bf16x8  __attribute__((ext_vector_type(8)));
typedef bf16  bf16x16 __attribute__((ext_vector_type(16)));
typedef float f32x4   __attribute__((ext_vector_type(4)));
typedef float f32x8   __attribute__((ext_vector_type(8)));

constexpr int B_  = 16;
constexpr int C_  = 1024;
constexpr int T_  = 1024;
constexpr int TE_ = 1024;
constexpr float SCALE_W = 0.70710678118654752440f; // sqrt(0.5)

constexpr int BK  = 32;   // K per WMMA step (bf16)
constexpr int LDA = 48;   // padded LDS stride (elems): 96B rows, 16B aligned
constexpr int SS  = 1032; // scores row stride (f32), de-conflicts banks

// LDS layout (bytes): 4 staging buffers (A0,B0,A1,B1) then (K2 only) scores.
constexpr int STG        = 64 * LDA * 2;  // 6144 B per 64x32 bf16 tile
constexpr int OFF_A0     = 0;
constexpr int OFF_B0     = STG;
constexpr int OFF_A1     = 2 * STG;
constexpr int OFF_B1     = 3 * STG;
constexpr int OFF_SCORES = 4 * STG;       // 24576
constexpr size_t LDS_GEMM = 4 * STG;                       // 24 KB
constexpr size_t LDS_SCOR = 4 * STG + (size_t)64 * SS * 4; // ~289 KB (< 320 KB WGP LDS)

// ---------------------------------------------------------------------------
// WMMA wrapper: D = A(16x32 bf16) * B(32x16 bf16) + C(f32)
// ---------------------------------------------------------------------------
__device__ __forceinline__ f32x8 wmma_bf16(bf16x16 a, bf16x16 b, f32x8 c) {
  return __builtin_amdgcn_wmma_f32_16x16x32_bf16(
      /*neg_a=*/false, a, /*neg_b=*/false, b,
      /*c_mod=*/(short)0, c, /*reuse_a=*/false, /*reuse_b=*/false);
}

// A fragment (ISA 7.12.2, 16-bit A 16x32): lane(laneLo,laneHi) holds row
// M=laneLo, K = {8*laneHi+0..7} in v0-3 and {16+8*laneHi+0..7} in v4-7.
__device__ __forceinline__ bf16x16 load_a_frag(const bf16* ldsA, int row, int laneHi) {
  const bf16x8 lo = *(const bf16x8*)(ldsA + row * LDA + 8 * laneHi);
  const bf16x8 hi = *(const bf16x8*)(ldsA + row * LDA + 16 + 8 * laneHi);
  bf16x16 a;
#pragma unroll
  for (int i = 0; i < 8; ++i) { a[i] = lo[i]; a[8 + i] = hi[i]; }
  return a;
}

// B fragment (32x16 bf16): lane holds column N=laneLo, K = {8*laneHi+0..7} then {16+8*laneHi+0..7}.
// LDS tile is stored [n][k] (k contiguous) -> two 16B chunks per lane, with a 16B gap between them.
__device__ __forceinline__ bf16x16 load_b_frag(const bf16* ldsBT, int col, int laneHi) {
  const bf16x8 lo = *(const bf16x8*)(ldsBT + col * LDA + 8 * laneHi);
  const bf16x8 hi = *(const bf16x8*)(ldsBT + col * LDA + 16 + 8 * laneHi);
  bf16x16 b;
#pragma unroll
  for (int i = 0; i < 8; ++i) { b[i] = lo[i]; b[8 + i] = hi[i]; }
  return b;
}

// ---------------------------------------------------------------------------
// Double-buffered async GEMM core. Arows: 64 M-rows x K (stride strideA),
// Brows: 64 N-rows x K (stride strideB), both K-contiguous bf16, 16B aligned.
// Produces two 16x16 f32 accumulators per wave (wave tile 16M x 32N).
// ---------------------------------------------------------------------------
__device__ __forceinline__ void gemm_core_async(
    const bf16* __restrict__ Arows, int strideA,
    const bf16* __restrict__ Brows, int strideB,
    int K, char* smem, unsigned smemOff,
    int tid, int m_w, int n_w, int laneLo, int laneHi,
    f32x8& acc0, f32x8& acc1)
{
  bf16* bufA[2] = { (bf16*)(smem + OFF_A0), (bf16*)(smem + OFF_A1) };
  bf16* bufB[2] = { (bf16*)(smem + OFF_B0), (bf16*)(smem + OFF_B1) };
  const unsigned offA[2] = { smemOff + OFF_A0, smemOff + OFF_A1 };
  const unsigned offB[2] = { smemOff + OFF_B0, smemOff + OFF_B1 };

  const int r  = tid >> 2;        // 0..63: tile row (M for A, N for B)
  const int kc = (tid & 3) * 8;   // 0,8,16,24: k-chunk (16 bytes)
  const unsigned ldsOff = (unsigned)(r * LDA + kc) * 2;

  // Each thread copies one 16B chunk of A and one of B per stage: the whole
  // 64x32 tile lands in LDS with no VGPR round-trip, tracked by ASYNCcnt.
  auto issue = [&](int buf, int kt) {
    asm volatile("global_load_async_to_lds_b128 %0, %1, off"
                 :: "v"(offA[buf] + ldsOff),
                    "v"(Arows + (size_t)r * strideA + kt + kc)
                 : "memory");
    asm volatile("global_load_async_to_lds_b128 %0, %1, off"
                 :: "v"(offB[buf] + ldsOff),
                    "v"(Brows + (size_t)r * strideB + kt + kc)
                 : "memory");
  };

  issue(0, 0);
  int cur = 0;
  for (int kt = 0; kt < K; kt += BK) {
    if (kt + BK < K) {
      issue(cur ^ 1, kt + BK);                              // prefetch next tile
      asm volatile("s_wait_asynccnt %0" :: "i"(2) : "memory"); // cur's 2 done (in-order)
    } else {
      asm volatile("s_wait_asynccnt %0" :: "i"(0) : "memory");
    }
    __syncthreads();  // cur tile visible to all waves
    const bf16x16 a  = load_a_frag(bufA[cur], m_w + laneLo, laneHi);
    const bf16x16 b0 = load_b_frag(bufB[cur], n_w + laneLo, laneHi);
    const bf16x16 b1 = load_b_frag(bufB[cur], n_w + 16 + laneLo, laneHi);
    acc0 = wmma_bf16(a, b0, acc0);
    acc1 = wmma_bf16(a, b1, acc1);
    __syncthreads();  // all waves done with cur before it is re-staged
    cur ^= 1;
  }
}

// ---------------------------------------------------------------------------
// K0a: plain f32 -> bf16 cast (8 elems / thread)
// ---------------------------------------------------------------------------
__global__ void cmsa_cast_f32_bf16(const float* __restrict__ src,
                                   bf16* __restrict__ dst, long long n8) {
  const long long i = (long long)blockIdx.x * blockDim.x + threadIdx.x;
  if (i >= n8) return;
  const f32x4* s = (const f32x4*)src + i * 2;
  const f32x4 v0 = s[0], v1 = s[1];
  bf16x8 o;
#pragma unroll
  for (int r = 0; r < 4; ++r) { o[r] = (bf16)v0[r]; o[4 + r] = (bf16)v1[r]; }
  *((bf16x8*)dst + i) = o;
}

// ---------------------------------------------------------------------------
// K0b: transposing cast. src f32 (R x Cc) row-major -> dst bf16 (Cc x R),
// per-batch via blockIdx.z. 64x64 tiles through LDS, coalesced both sides.
// ---------------------------------------------------------------------------
__global__ void cmsa_castT_f32_bf16(const float* __restrict__ src,
                                    bf16* __restrict__ dst, int R, int Cc) {
  __shared__ float tile[64][68];  // 68-f32 stride: 16B-aligned rows, no conflicts
  const int tid = threadIdx.x;
  const size_t bOff = (size_t)blockIdx.z * R * Cc;
  const int r0 = blockIdx.y * 64;
  const int c0 = blockIdx.x * 64;
#pragma unroll
  for (int p = 0; p < 4; ++p) {
    const int r  = p * 16 + (tid >> 4);
    const int c4 = (tid & 15) * 4;
    *(f32x4*)&tile[r][c4] =
        *(const f32x4*)(src + bOff + (size_t)(r0 + r) * Cc + c0 + c4);
  }
  __syncthreads();
#pragma unroll
  for (int p = 0; p < 2; ++p) {
    const int n  = p * 32 + (tid >> 3);  // output row (= src col)
    const int rc = (tid & 7) * 8;        // output col chunk (= src rows)
    bf16x8 o;
#pragma unroll
    for (int j = 0; j < 8; ++j) o[j] = (bf16)tile[rc + j][n];
    *(bf16x8*)(dst + bOff + (size_t)(c0 + n) * R + r0 + rc) = o;
  }
}

// ---------------------------------------------------------------------------
// K1: target = (base + W @ X + bias) * SCALE, stored transposed bf16 (B,T,C).
// A = W (C x C), B^T rows = X^T (T x C). Grid (T/64, C/64, B), 256 thr.
// ---------------------------------------------------------------------------
__global__ void cmsa_gemm1_target(const bf16* __restrict__ Wbf,
                                  const bf16* __restrict__ XT,
                                  const float* __restrict__ base,
                                  const float* __restrict__ bias,
                                  bf16* __restrict__ targetT) {
  extern __shared__ char smem[];
  const unsigned smemOff = (unsigned)(size_t)smem;  // wave-relative LDS offset
  const int tid = threadIdx.x;
  const int wave = tid >> 5, lane = tid & 31;
  const int laneLo = lane & 15, laneHi = lane >> 4;
  const int m_w = (wave & 3) * 16;   // c offset in tile
  const int n_w = (wave >> 2) * 32;  // t offset in tile
  const int n0 = blockIdx.x * 64;    // t tile
  const int m0 = blockIdx.y * 64;    // c tile
  const int b  = blockIdx.z;

  f32x8 acc0 = {}, acc1 = {};
  gemm_core_async(Wbf + (size_t)m0 * C_, C_,
                  XT + (size_t)b * T_ * C_ + (size_t)n0 * C_, C_,
                  C_, smem, smemOff, tid, m_w, n_w, laneLo, laneHi, acc0, acc1);

  // D mapping: N=laneLo (t), M=r+8*laneHi (c).
  const int cBase = m0 + m_w + 8 * laneHi;
#pragma unroll
  for (int ns = 0; ns < 2; ++ns) {
    const f32x8 acc = ns ? acc1 : acc0;
    const int t = n0 + n_w + ns * 16 + laneLo;
    bf16x8 pack;
#pragma unroll
    for (int r = 0; r < 8; ++r) {
      const int c = cBase + r;
      const float v = (base[((size_t)b * C_ + c) * T_ + t] + acc[r] + bias[c]) * SCALE_W;
      pack[r] = (bf16)v;
    }
    *(bf16x8*)(targetT + ((size_t)b * T_ + t) * C_ + cBase) = pack;
  }
}

// ---------------------------------------------------------------------------
// K2: scores = targetT @ enc_top^T, fused softmax over e; scores in LDS
// (64 t-rows x Te=1024 f32 = 264KB -- only possible with CDNA5's 320KB WGP LDS).
// Grid (T/64, B), 256 thr.
// ---------------------------------------------------------------------------
__global__ void cmsa_scores_softmax(const bf16* __restrict__ targetT,
                                    const bf16* __restrict__ topbf,
                                    float* __restrict__ attn_out,
                                    bf16* __restrict__ attn_bf) {
  extern __shared__ char smem[];
  const unsigned smemOff = (unsigned)(size_t)smem;
  float* scores = (float*)(smem + OFF_SCORES);
  const int tid = threadIdx.x;
  const int wave = tid >> 5, lane = tid & 31;
  const int laneLo = lane & 15, laneHi = lane >> 4;
  const int m_w = (wave & 3) * 16;
  const int n_w = (wave >> 2) * 32;
  const int t0 = blockIdx.x * 64;
  const int b  = blockIdx.y;

  const bf16* Asrc = targetT + (size_t)b * T_ * C_;   // (T x C), t rows
  const bf16* Ksrc = topbf   + (size_t)b * TE_ * C_;  // (Te x C), e rows

  for (int et = 0; et < TE_ / 64; ++et) {
    const int n0 = et * 64;
    f32x8 acc0 = {}, acc1 = {};
    gemm_core_async(Asrc + (size_t)t0 * C_, C_,
                    Ksrc + (size_t)n0 * C_, C_,
                    C_, smem, smemOff, tid, m_w, n_w, laneLo, laneHi, acc0, acc1);
    const int tRow = m_w + 8 * laneHi;
#pragma unroll
    for (int ns = 0; ns < 2; ++ns) {
      const f32x8 acc = ns ? acc1 : acc0;
      const int e = n0 + n_w + ns * 16 + laneLo;
#pragma unroll
      for (int r = 0; r < 8; ++r) scores[(tRow + r) * SS + e] = acc[r];
    }
  }
  __syncthreads();

  // Softmax: each wave owns 8 t-rows; lane covers e = lane + 32j.
  for (int rr = 0; rr < 8; ++rr) {
    const int t = wave * 8 + rr;
    float vals[TE_ / 32];
    float mx = -3.402823466e38f;
#pragma unroll
    for (int j = 0; j < TE_ / 32; ++j) {
      vals[j] = scores[t * SS + lane + 32 * j];
      mx = fmaxf(mx, vals[j]);
    }
#pragma unroll
    for (int off = 16; off > 0; off >>= 1) mx = fmaxf(mx, __shfl_xor(mx, off, 32));
    float sum = 0.f;
#pragma unroll
    for (int j = 0; j < TE_ / 32; ++j) { vals[j] = __expf(vals[j] - mx); sum += vals[j]; }
#pragma unroll
    for (int off = 16; off > 0; off >>= 1) sum += __shfl_xor(sum, off, 32);
    const float inv = 1.0f / sum;
    const size_t rowOff = ((size_t)b * T_ + t0 + t) * TE_;
#pragma unroll
    for (int j = 0; j < TE_ / 32; ++j) {
      const float a = vals[j] * inv;
      attn_out[rowOff + lane + 32 * j] = a;        // f32 output (tuple elem 2)
      attn_bf [rowOff + lane + 32 * j] = (bf16)a;  // bf16 feed for GEMM3
    }
  }
}

// ---------------------------------------------------------------------------
// K3: context(B,C,T) = transpose( attn(T x Te) @ enc_comb(Te x C) ).
// A = attn_bf (T x Te), B^T rows = enc_comb^T (C x Te). Grid (T/64, C/64, B).
// ---------------------------------------------------------------------------
__global__ void cmsa_context(const bf16* __restrict__ attn_bf,
                             const bf16* __restrict__ combT,
                             float* __restrict__ ctx_out) {
  extern __shared__ char smem[];
  const unsigned smemOff = (unsigned)(size_t)smem;
  const int tid = threadIdx.x;
  const int wave = tid >> 5, lane = tid & 31;
  const int laneLo = lane & 15, laneHi = lane >> 4;
  const int m_w = (wave & 3) * 16;   // t offset
  const int n_w = (wave >> 2) * 32;  // c offset
  const int m0 = blockIdx.x * 64;    // t tile
  const int n0 = blockIdx.y * 64;    // c tile
  const int b  = blockIdx.z;

  f32x8 acc0 = {}, acc1 = {};
  gemm_core_async(attn_bf + (size_t)b * T_ * TE_ + (size_t)m0 * TE_, TE_,
                  combT + (size_t)b * C_ * TE_ + (size_t)n0 * TE_, TE_,
                  TE_, smem, smemOff, tid, m_w, n_w, laneLo, laneHi, acc0, acc1);

  // D mapping: N=laneLo -> c, M=r+8*laneHi -> t; 8 consecutive t per lane.
  const int tBase = m0 + m_w + 8 * laneHi;
#pragma unroll
  for (int ns = 0; ns < 2; ++ns) {
    const f32x8 acc = ns ? acc1 : acc0;
    const int c = n0 + n_w + ns * 16 + laneLo;
    float* dst = ctx_out + ((size_t)b * C_ + c) * T_ + tBase;
    f32x4 lo, hi;
#pragma unroll
    for (int r = 0; r < 4; ++r) { lo[r] = acc[r]; hi[r] = acc[4 + r]; }
    *(f32x4*)dst       = lo;
    *(f32x4*)(dst + 4) = hi;
  }
}

// ---------------------------------------------------------------------------
extern "C" void kernel_launch(void* const* d_in, const int* in_sizes, int n_in,
                              void* d_out, int out_size, void* d_ws, size_t ws_size,
                              hipStream_t stream) {
  (void)in_sizes; (void)n_in; (void)out_size; (void)ws_size;
  const float* base = (const float*)d_in[0];  // (B,C,T)
  const float* xdec = (const float*)d_in[1];  // (B,C,T)
  const float* top  = (const float*)d_in[2];  // (B,Te,C)
  const float* comb = (const float*)d_in[3];  // (B,Te,C)
  const float* W    = (const float*)d_in[4];  // (C,C)
  const float* bias = (const float*)d_in[5];  // (C)

  float* ctx_out  = (float*)d_out;                          // (B,C,T)
  float* attn_out = (float*)d_out + (size_t)B_ * C_ * T_;   // (B,T,Te)

  // Workspace partition (bf16 elems), ~162 MB total.
  bf16* ws = (bf16*)d_ws;
  size_t off = 0;
  bf16* Wbf     = ws + off; off += (size_t)C_ * C_;        // (C,C)
  bf16* XT      = ws + off; off += (size_t)B_ * T_ * C_;   // (B,T,C)  = X^T
  bf16* targetT = ws + off; off += (size_t)B_ * T_ * C_;   // (B,T,C)
  bf16* topbf   = ws + off; off += (size_t)B_ * TE_ * C_;  // (B,Te,C)
  bf16* combT   = ws + off; off += (size_t)B_ * C_ * TE_;  // (B,C,Te) = comb^T
  bf16* attnbf  = ws + off; off += (size_t)B_ * T_ * TE_;  // (B,T,Te)

  auto cast = [&](const float* s, bf16* d, size_t n) {
    const long long n8 = (long long)(n / 8);
    cmsa_cast_f32_bf16<<<dim3((unsigned)((n8 + 255) / 256)), 256, 0, stream>>>(s, d, n8);
  };
  cast(W,   Wbf,   (size_t)C_ * C_);
  cast(top, topbf, (size_t)B_ * TE_ * C_);
  // X (B,C,T) -> X^T (B,T,C);  comb (B,Te,C) -> comb^T (B,C,Te)
  cmsa_castT_f32_bf16<<<dim3(T_ / 64, C_ / 64, B_), 256, 0, stream>>>(xdec, XT, C_, T_);
  cmsa_castT_f32_bf16<<<dim3(C_ / 64, TE_ / 64, B_), 256, 0, stream>>>(comb, combT, TE_, C_);

  cmsa_gemm1_target<<<dim3(T_ / 64, C_ / 64, B_), 256, LDS_GEMM, stream>>>(
      Wbf, XT, base, bias, targetT);

  cmsa_scores_softmax<<<dim3(T_ / 64, B_), 256, LDS_SCOR, stream>>>(
      targetT, topbf, attn_out, attnbf);

  cmsa_context<<<dim3(T_ / 64, C_ / 64, B_), 256, LDS_GEMM, stream>>>(
      attnbf, combT, ctx_out);
}
